// Pointnet2_Ssg_9998683865542
// MI455X (gfx1250) — matrix operand order repaired and checked by reading, not compiled, not run on hardware
//
#include <hip/hip_runtime.h>
#include <hip/hip_bf16.h>

typedef __bf16 bf16_t;
typedef bf16_t v16bf __attribute__((ext_vector_type(16)));
typedef bf16_t v8bf  __attribute__((ext_vector_type(8)));
typedef float  v8f   __attribute__((ext_vector_type(8)));

#define BN_EPS 1e-5f

// ---------------------------------------------------------------------------
// Transpose (B,3,N) -> (B,N,3)
// ---------------------------------------------------------------------------
__global__ void k_transpose_xyz(const float* __restrict__ xyz,
                                float* __restrict__ pts, int B, int N) {
  int i = blockIdx.x * blockDim.x + threadIdx.x;
  if (i >= B * N) return;
  int b = i / N, n = i % N;
  const float* s = xyz + (size_t)b * 3 * N;
  float* d = pts + (size_t)i * 3;
  d[0] = s[n]; d[1] = s[N + n]; d[2] = s[2 * N + n];
}

// ---------------------------------------------------------------------------
// Farthest point sampling: one block per batch, dist[] resident in LDS.
// ---------------------------------------------------------------------------
__global__ __launch_bounds__(512) void k_fps(const float* __restrict__ pts,
                                             int* __restrict__ idxOut,
                                             int N, int npoint) {
  __shared__ float dist[8192];
  __shared__ float redV[512];
  __shared__ int   redI[512];
  __shared__ int   s_far;
  int b = blockIdx.x, t = threadIdx.x, nt = blockDim.x;
  const float* P = pts + (size_t)b * N * 3;
  for (int i = t; i < N; i += nt) dist[i] = 1e10f;
  if (t == 0) s_far = 0;
  __syncthreads();
  for (int it = 0; it < npoint; ++it) {
    int far = s_far;
    if (t == 0) idxOut[b * npoint + it] = far;
    float cx = P[far * 3 + 0], cy = P[far * 3 + 1], cz = P[far * 3 + 2];
    float bestV = -1.0f; int bestI = 0;
    for (int i = t; i < N; i += nt) {
      float dx = P[i * 3 + 0] - cx;
      float dy = P[i * 3 + 1] - cy;
      float dz = P[i * 3 + 2] - cz;
      float d = dx * dx + dy * dy + dz * dz;
      float m = fminf(dist[i], d);
      dist[i] = m;
      if (m > bestV) { bestV = m; bestI = i; }
    }
    redV[t] = bestV; redI[t] = bestI;
    __syncthreads();
    for (int off = nt >> 1; off > 0; off >>= 1) {
      if (t < off) {
        bool take = (redV[t + off] > redV[t]) ||
                    (redV[t + off] == redV[t] && redI[t + off] < redI[t]);
        if (take) { redV[t] = redV[t + off]; redI[t] = redI[t + off]; }
      }
      __syncthreads();
    }
    if (t == 0) s_far = redI[0];
    __syncthreads();
  }
}

// ---------------------------------------------------------------------------
// Gather sampled centroids: (B,N,3) x (B,S) -> (B,S,3)
// ---------------------------------------------------------------------------
__global__ void k_gather_xyz(const float* __restrict__ pts,
                             const int* __restrict__ idx,
                             float* __restrict__ nxyz, int N, int S, int B) {
  int i = blockIdx.x * blockDim.x + threadIdx.x;
  if (i >= B * S) return;
  int b = i / S;
  int id = idx[i];
  const float* p = pts + ((size_t)b * N + id) * 3;
  float* d = nxyz + (size_t)i * 3;
  d[0] = p[0]; d[1] = p[1]; d[2] = p[2];
}

// ---------------------------------------------------------------------------
// Ball query + grouping: one wave per (b,s). First `nsample` in-radius
// indices in ascending order (== sort-based reference), remainder filled with
// the first neighbor; rows written as bf16 padded to Kpad for WMMA.
// ---------------------------------------------------------------------------
__global__ __launch_bounds__(32) void k_ball_group(
    const float* __restrict__ xyz, const float* __restrict__ nxyz,
    const float* __restrict__ feat, bf16_t* __restrict__ A0,
    int N, int S, int nsample, float r2, int Cf, int Kpad) {
  int bs = blockIdx.x;
  int b = bs / S;
  int lane = threadIdx.x;
  __shared__ int list[64];
  const float* nx = nxyz + (size_t)bs * 3;
  float cx = nx[0], cy = nx[1], cz = nx[2];
  int count = 0;
  for (int base = 0; base < N && count < nsample; base += 32) {
    int i = base + lane;
    bool ok = false;
    if (i < N) {
      const float* p = xyz + ((size_t)b * N + i) * 3;
      float dx = p[0] - cx, dy = p[1] - cy, dz = p[2] - cz;
      ok = (dx * dx + dy * dy + dz * dz) <= r2;
    }
    unsigned mask = (unsigned)__ballot(ok);
    int pre = __popc(mask & ((1u << lane) - 1u));
    int slot = count + pre;
    if (ok && slot < nsample) list[slot] = i;
    count += __popc(mask);
  }
  __syncthreads();
  int cnt = count < nsample ? count : nsample;
  int first = (cnt > 0) ? list[0] : (N - 1);
  for (int k = lane; k < nsample; k += 32) {
    int id = (k < cnt) ? list[k] : first;
    bf16_t* out = A0 + ((size_t)bs * nsample + k) * Kpad;
    const float* p = xyz + ((size_t)b * N + id) * 3;
    out[0] = (bf16_t)(p[0] - cx);
    out[1] = (bf16_t)(p[1] - cy);
    out[2] = (bf16_t)(p[2] - cz);
    int c = 3;
    if (feat) {
      const float* f = feat + ((size_t)b * N + id) * Cf;
      for (int j = 0; j < Cf; ++j) out[3 + j] = (bf16_t)f[j];
      c = 3 + Cf;
    }
    for (; c < Kpad; ++c) out[c] = (bf16_t)0.0f;
  }
}

// ---------------------------------------------------------------------------
// group_all for SA3: concat(l2_xyz, l2_pts) per row, pad K to Kpad.
// ---------------------------------------------------------------------------
__global__ void k_group_all(const float* __restrict__ xyz2,
                            const float* __restrict__ pts2,
                            bf16_t* __restrict__ A, int B, int N, int Cf,
                            int Kpad) {
  int i = blockIdx.x * blockDim.x + threadIdx.x;
  if (i >= B * N) return;
  bf16_t* out = A + (size_t)i * Kpad;
  const float* p = xyz2 + (size_t)i * 3;
  out[0] = (bf16_t)p[0]; out[1] = (bf16_t)p[1]; out[2] = (bf16_t)p[2];
  const float* f = pts2 + (size_t)i * Cf;
  for (int j = 0; j < Cf; ++j) out[3 + j] = (bf16_t)f[j];
  for (int c = 3 + Cf; c < Kpad; ++c) out[c] = (bf16_t)0.0f;
}

// ---------------------------------------------------------------------------
// Convert f32 weights (O x C) -> bf16 (O x Kpad), zero padded.
// ---------------------------------------------------------------------------
__global__ void k_conv_w(const float* __restrict__ W, bf16_t* __restrict__ Wb,
                         int O, int C, int Kpad) {
  int i = blockIdx.x * blockDim.x + threadIdx.x;
  if (i >= O * Kpad) return;
  int o = i / Kpad, k = i % Kpad;
  Wb[i] = (k < C) ? (bf16_t)W[(size_t)o * C + k] : (bf16_t)0.0f;
}

// ---------------------------------------------------------------------------
// Fragment helpers (ISA 16-bit A 16x32 / B 32x16 wave32 layouts).
// ---------------------------------------------------------------------------
__device__ __forceinline__ v16bf load_a_frag(const bf16_t* Ap, int k,
                                             int half) {
  v8bf lo = *(const v8bf*)(Ap + k + half * 8);
  v8bf hi = *(const v8bf*)(Ap + k + 16 + half * 8);
  return __builtin_shufflevector(lo, hi, 0, 1, 2, 3, 4, 5, 6, 7, 8, 9, 10, 11,
                                 12, 13, 14, 15);
}
__device__ __forceinline__ v16bf load_b_frag(const bf16_t* Bp, int k) {
  v8bf lo = *(const v8bf*)(Bp + k);
  v8bf hi = *(const v8bf*)(Bp + k + 16);
  return __builtin_shufflevector(lo, hi, 0, 1, 2, 3, 4, 5, 6, 7, 8, 9, 10, 11,
                                 12, 13, 14, 15);
}
__device__ __forceinline__ void bn_params(const float* bias,
                                          const float* gamma,
                                          const float* beta, const float* mu,
                                          const float* var, int c, int fuse,
                                          float& bb, float& s, float& off) {
  bb = bias[c];
  s = 1.0f; off = 0.0f;
  if (fuse) {
    float sc = gamma[c] * rsqrtf(var[c] + BN_EPS);
    s = sc;
    off = beta[c] - mu[c] * sc;
  }
}

// ---------------------------------------------------------------------------
// 2x2-tile WMMA GEMM: each wave computes a 32x32 output tile (4 WMMAs per
// k-step from 8 b128 loads -> 2 loads/WMMA). Requires M%32==0, O%32==0.
// out[M,O] = A[M,Ka] @ W^T (+bias, optional fused BN+ReLU).
// ---------------------------------------------------------------------------
__global__ __launch_bounds__(32) void k_gemm_wmma_2x2(
    const bf16_t* __restrict__ A, const bf16_t* __restrict__ Wb,
    const float* __restrict__ bias, const float* __restrict__ gamma,
    const float* __restrict__ beta, const float* __restrict__ mu,
    const float* __restrict__ var, bf16_t* __restrict__ outB,
    float* __restrict__ outF, int M, int Ka, int O, int fuse) {
  int lane = threadIdx.x;
  int half = lane >> 4, r = lane & 15;
  int m0 = blockIdx.x * 32, n0 = blockIdx.y * 32;
  const bf16_t* Ap0 = A + (size_t)(m0 + r) * Ka;
  const bf16_t* Ap1 = A + (size_t)(m0 + 16 + r) * Ka;
  const bf16_t* Bp0 = Wb + (size_t)(n0 + r) * Ka + half * 8;
  const bf16_t* Bp1 = Wb + (size_t)(n0 + 16 + r) * Ka + half * 8;
  v8f acc00 = {}, acc01 = {}, acc10 = {}, acc11 = {};
  for (int k = 0; k < Ka; k += 32) {
    __builtin_prefetch(Ap0 + k + 128, 0, 0);
    __builtin_prefetch(Ap1 + k + 128, 0, 0);
    __builtin_prefetch(Bp0 + k + 128, 0, 0);
    __builtin_prefetch(Bp1 + k + 128, 0, 0);
    v16bf a0 = load_a_frag(Ap0, k, half);
    v16bf a1 = load_a_frag(Ap1, k, half);
    v16bf b0 = load_b_frag(Bp0, k);
    v16bf b1 = load_b_frag(Bp1, k);
    acc00 = __builtin_amdgcn_wmma_f32_16x16x32_bf16(false, a0, false, b0,
                                                    (short)0, acc00, false,
                                                    false);
    acc01 = __builtin_amdgcn_wmma_f32_16x16x32_bf16(false, a0, false, b1,
                                                    (short)0, acc01, false,
                                                    false);
    acc10 = __builtin_amdgcn_wmma_f32_16x16x32_bf16(false, a1, false, b0,
                                                    (short)0, acc10, false,
                                                    false);
    acc11 = __builtin_amdgcn_wmma_f32_16x16x32_bf16(false, a1, false, b1,
                                                    (short)0, acc11, false,
                                                    false);
  }
  int c0 = n0 + r, c1 = n0 + 16 + r;
  float bb0, s0, o0, bb1, s1, o1;
  bn_params(bias, gamma, beta, mu, var, c0, fuse, bb0, s0, o0);
  bn_params(bias, gamma, beta, mu, var, c1, fuse, bb1, s1, o1);
#pragma unroll
  for (int v = 0; v < 8; ++v) {
    int row0 = m0 + v + half * 8;
    int row1 = row0 + 16;
    float x00 = acc00[v] + bb0, x01 = acc01[v] + bb1;
    float x10 = acc10[v] + bb0, x11 = acc11[v] + bb1;
    if (fuse) {
      x00 = fmaxf(x00 * s0 + o0, 0.0f);
      x01 = fmaxf(x01 * s1 + o1, 0.0f);
      x10 = fmaxf(x10 * s0 + o0, 0.0f);
      x11 = fmaxf(x11 * s1 + o1, 0.0f);
    }
    size_t o00 = (size_t)row0 * O + c0, o01 = (size_t)row0 * O + c1;
    size_t o10 = (size_t)row1 * O + c0, o11 = (size_t)row1 * O + c1;
    if (outB) {
      outB[o00] = (bf16_t)x00; outB[o01] = (bf16_t)x01;
      outB[o10] = (bf16_t)x10; outB[o11] = (bf16_t)x11;
    }
    if (outF) {
      outF[o00] = x00; outF[o01] = x01;
      outF[o10] = x10; outF[o11] = x11;
    }
  }
}

// ---------------------------------------------------------------------------
// 1x1-tile WMMA GEMM (for the M=16 FC head).
// ---------------------------------------------------------------------------
__global__ __launch_bounds__(32) void k_gemm_wmma(
    const bf16_t* __restrict__ A, const bf16_t* __restrict__ Wb,
    const float* __restrict__ bias, const float* __restrict__ gamma,
    const float* __restrict__ beta, const float* __restrict__ mu,
    const float* __restrict__ var, bf16_t* __restrict__ outB,
    float* __restrict__ outF, int M, int Ka, int O, int fuse) {
  int lane = threadIdx.x;
  int half = lane >> 4, r = lane & 15;
  int m0 = blockIdx.x * 16, n0 = blockIdx.y * 16;
  const bf16_t* Ap = A + (size_t)(m0 + r) * Ka;
  const bf16_t* Bp = Wb + (size_t)(n0 + r) * Ka + half * 8;
  v8f acc = {};
  for (int k = 0; k < Ka; k += 32) {
    __builtin_prefetch(Ap + k + 128, 0, 0);
    __builtin_prefetch(Bp + k + 128, 0, 0);
    v16bf a = load_a_frag(Ap, k, half);
    v16bf bm = load_b_frag(Bp, k);
    acc = __builtin_amdgcn_wmma_f32_16x16x32_bf16(false, a, false, bm,
                                                  (short)0, acc, false, false);
  }
  int c = n0 + r;
  float bb, s, off;
  bn_params(bias, gamma, beta, mu, var, c, fuse, bb, s, off);
#pragma unroll
  for (int v = 0; v < 8; ++v) {
    int row = m0 + v + half * 8;
    float x = acc[v] + bb;
    if (fuse) x = fmaxf(x * s + off, 0.0f);
    size_t o = (size_t)row * O + c;
    if (outB) outB[o] = (bf16_t)x;
    if (outF) outF[o] = x;
  }
}

// ---------------------------------------------------------------------------
// Max pool over G consecutive rows per group (bf16 in, f32 and/or bf16 out).
// ---------------------------------------------------------------------------
__global__ void k_maxpool(const bf16_t* __restrict__ in,
                          float* __restrict__ outF, bf16_t* __restrict__ outB,
                          int groups, int G, int O) {
  int i = blockIdx.x * blockDim.x + threadIdx.x;
  if (i >= groups * O) return;
  int g = i / O, c = i % O;
  const bf16_t* p = in + (size_t)g * G * O + c;
  float m = -INFINITY;
  for (int k = 0; k < G; ++k) m = fmaxf(m, (float)p[(size_t)k * O]);
  if (outF) outF[i] = m;
  if (outB) outB[i] = (bf16_t)m;
}

// ---------------------------------------------------------------------------
// Host side
// ---------------------------------------------------------------------------
struct LayerP { const float *W, *b, *g, *be, *mu, *var; };

static LayerP layer_at(void* const* d_in, int base) {
  LayerP L;
  L.W = (const float*)d_in[base + 0];
  L.b = (const float*)d_in[base + 1];
  L.g = (const float*)d_in[base + 2];
  L.be = (const float*)d_in[base + 3];
  L.mu = (const float*)d_in[base + 4];
  L.var = (const float*)d_in[base + 5];
  return L;
}

extern "C" void kernel_launch(void* const* d_in, const int* in_sizes, int n_in,
                              void* d_out, int out_size, void* d_ws,
                              size_t ws_size, hipStream_t stream) {
  (void)in_sizes; (void)n_in; (void)out_size; (void)ws_size;
  const int B = 16, N = 8192;
  const float* xyz = (const float*)d_in[0];
  // input order: xyz, sa1(L0..L2), sa2(L0..L2), sa3(L0..L2), fc1, fc2
  LayerP s1[3] = {layer_at(d_in, 1), layer_at(d_in, 7), layer_at(d_in, 13)};
  LayerP s2[3] = {layer_at(d_in, 19), layer_at(d_in, 25), layer_at(d_in, 31)};
  LayerP s3[3] = {layer_at(d_in, 37), layer_at(d_in, 43), layer_at(d_in, 49)};
  LayerP fc1 = layer_at(d_in, 55);
  LayerP fc2 = layer_at(d_in, 61);

  // -------- workspace arena --------
  size_t off = 0;
  char* base = (char*)d_ws;
  auto alloc = [&](size_t bytes) {
    char* p = base + off;
    off = (off + bytes + 255) & ~(size_t)255;
    return p;
  };
  float* pts    = (float*)alloc((size_t)B * N * 3 * 4);
  int*   idx1   = (int*)alloc((size_t)B * 512 * 4);
  float* nxyz1  = (float*)alloc((size_t)B * 512 * 3 * 4);
  int*   idx2   = (int*)alloc((size_t)B * 128 * 4);
  float* nxyz2  = (float*)alloc((size_t)B * 128 * 3 * 4);
  float* l1pts  = (float*)alloc((size_t)B * 512 * 128 * 4);
  float* l2pts  = (float*)alloc((size_t)B * 128 * 256 * 4);
  bf16_t* l3bf  = (bf16_t*)alloc((size_t)B * 1024 * 2);
  bf16_t* fcbf  = (bf16_t*)alloc((size_t)B * 512 * 2);
  bf16_t* Wb    = (bf16_t*)alloc((size_t)1024 * 1024 * 2);
  bf16_t* X     = (bf16_t*)alloc((size_t)262144 * 128 * 2);  // 64 MB ping
  bf16_t* Y     = (bf16_t*)alloc((size_t)262144 * 128 * 2);  // 64 MB pong

  auto cdiv = [](int a, int b) { return (a + b - 1) / b; };
  auto conv_w = [&](const float* W, int O, int C, int Kp) {
    k_conv_w<<<cdiv(O * Kp, 256), 256, 0, stream>>>(W, Wb, O, C, Kp);
  };
  auto gemm2 = [&](const bf16_t* A, const LayerP& L, bf16_t* oB, float* oF,
                   int M, int Ka, int O, int fuse) {
    dim3 g(M / 32, O / 32);
    k_gemm_wmma_2x2<<<g, 32, 0, stream>>>(A, Wb, L.b, L.g, L.be, L.mu, L.var,
                                          oB, oF, M, Ka, O, fuse);
  };
  auto gemm1 = [&](const bf16_t* A, const LayerP& L, bf16_t* oB, float* oF,
                   int M, int Ka, int O, int fuse) {
    dim3 g(M / 16, O / 16);
    k_gemm_wmma<<<g, 32, 0, stream>>>(A, Wb, L.b, L.g, L.be, L.mu, L.var, oB,
                                      oF, M, Ka, O, fuse);
  };

  // -------- stage 0: layout --------
  k_transpose_xyz<<<cdiv(B * N, 256), 256, 0, stream>>>(xyz, pts, B, N);

  // -------- SA1: npoint=512, r=0.2, ns=32, 3->64->64->128 --------
  k_fps<<<B, 512, 0, stream>>>(pts, idx1, N, 512);
  k_gather_xyz<<<cdiv(B * 512, 256), 256, 0, stream>>>(pts, idx1, nxyz1, N,
                                                       512, B);
  k_ball_group<<<B * 512, 32, 0, stream>>>(pts, nxyz1, nullptr, X, N, 512, 32,
                                           0.04f, 0, 32);
  int M1 = B * 512 * 32;  // 262144
  conv_w(s1[0].W, 64, 3, 32);
  gemm2(X, s1[0], Y, nullptr, M1, 32, 64, 1);
  conv_w(s1[1].W, 64, 64, 64);
  gemm2(Y, s1[1], X, nullptr, M1, 64, 64, 1);
  conv_w(s1[2].W, 128, 64, 64);
  gemm2(X, s1[2], Y, nullptr, M1, 64, 128, 1);
  k_maxpool<<<cdiv(B * 512 * 128, 256), 256, 0, stream>>>(Y, l1pts, nullptr,
                                                          B * 512, 32, 128);

  // -------- SA2: npoint=128, r=0.4, ns=64, 131->128->128->256 --------
  k_fps<<<B, 512, 0, stream>>>(nxyz1, idx2, 512, 128);
  k_gather_xyz<<<cdiv(B * 128, 256), 256, 0, stream>>>(nxyz1, idx2, nxyz2, 512,
                                                       128, B);
  k_ball_group<<<B * 128, 32, 0, stream>>>(nxyz1, nxyz2, l1pts, X, 512, 128,
                                           64, 0.16f, 128, 160);
  int M2 = B * 128 * 64;  // 131072
  conv_w(s2[0].W, 128, 131, 160);
  gemm2(X, s2[0], Y, nullptr, M2, 160, 128, 1);
  conv_w(s2[1].W, 128, 128, 128);
  gemm2(Y, s2[1], X, nullptr, M2, 128, 128, 1);
  conv_w(s2[2].W, 256, 128, 128);
  gemm2(X, s2[2], Y, nullptr, M2, 128, 256, 1);
  k_maxpool<<<cdiv(B * 128 * 256, 256), 256, 0, stream>>>(Y, l2pts, nullptr,
                                                          B * 128, 64, 256);

  // -------- SA3 (group all): 259->256->512->1024, last layer no BN/ReLU ----
  k_group_all<<<cdiv(B * 128, 256), 256, 0, stream>>>(nxyz2, l2pts, X, B, 128,
                                                      256, 288);
  int M3 = B * 128;  // 2048
  conv_w(s3[0].W, 256, 259, 288);
  gemm2(X, s3[0], Y, nullptr, M3, 288, 256, 1);
  conv_w(s3[1].W, 512, 256, 256);
  gemm2(Y, s3[1], X, nullptr, M3, 256, 512, 1);
  conv_w(s3[2].W, 1024, 512, 512);
  gemm2(X, s3[2], Y, nullptr, M3, 512, 1024, 0);  // remove_last: bias only
  k_maxpool<<<cdiv(B * 1024, 256), 256, 0, stream>>>(Y, nullptr, l3bf, B, 128,
                                                     1024);

  // -------- FC head (M=16) --------
  conv_w(fc1.W, 512, 1024, 1024);
  gemm1(l3bf, fc1, fcbf, nullptr, 16, 1024, 512, 1);
  conv_w(fc2.W, 256, 512, 512);
  gemm1(fcbf, fc2, nullptr, (float*)d_out, 16, 512, 256, 1);
}
